// STFullAttention_52415780880836
// MI455X (gfx1250) — matrix operand order repaired and checked by reading, not compiled, not run on hardware
//
#include <hip/hip_runtime.h>

// ---------------------------------------------------------------------------
// ST full attention for MI455X (gfx1250, wave32, WMMA).
// Pass 1 (temporal) and pass 2 (spatial, masked) share one attention core.
// Shapes: B=2, N=128, L=128, H=8, E=64. 1 block = 1 [128x128]x[128x64] attn.
// ---------------------------------------------------------------------------

typedef __attribute__((ext_vector_type(16))) __bf16          v16bf;
typedef __attribute__((ext_vector_type(8)))  float           v8f;
typedef __attribute__((ext_vector_type(4)))  float           v4f;
typedef __attribute__((ext_vector_type(4)))  unsigned int    v4u;
typedef __attribute__((ext_vector_type(4)))  unsigned short  v4us;

// float -> bf16 bits, round-to-nearest-even
__device__ __forceinline__ unsigned short f2bf(float x) {
    union { float f; unsigned int u; } v; v.f = x;
    unsigned int r = v.u + 0x7FFFu + ((v.u >> 16) & 1u);
    return (unsigned short)(r >> 16);
}

// A-fragment (16x32 bf16): lane(lo,hi) holds row M=mbase+lo,
// K chunks [kbase+hi*8 .. +7] and [kbase+16+hi*8 .. +7]  (two 16B LDS loads).
__device__ __forceinline__ v16bf ld_a_frag(const unsigned short* p, int ld,
                                           int mbase, int kbase, int lo, int hi) {
    const unsigned short* r = p + (mbase + lo) * ld + kbase + hi * 8;
    union { v4u q[2]; v16bf v; } t;
    t.q[0] = *(const v4u*)(r);
    t.q[1] = *(const v4u*)(r + 16);
    return t.v;
}

// B-fragment (32x16 bf16): lane(lo,hi) holds column N=nbase+lo,
// K chunks [kbase+hi*8 .. +7] and [kbase+16+hi*8 .. +7]  (two 16B LDS loads).
__device__ __forceinline__ v16bf ld_b_frag(const unsigned short* p, int ld,
                                           int nbase, int kbase, int lo, int hi) {
    const unsigned short* r = p + (nbase + lo) * ld + kbase + hi * 8;
    union { v4u q[2]; v16bf v; } t;
    t.q[0] = *(const v4u*)(r);
    t.q[1] = *(const v4u*)(r + 16);
    return t.v;
}

__device__ __forceinline__ v8f wmma_bf16(v16bf a, v16bf b, v8f c) {
    // D = A x B + C, f32 accumulate  -> v_wmma_f32_16x16x32_bf16
    return __builtin_amdgcn_wmma_f32_16x16x32_bf16(
        /*neg_a=*/false, a, /*neg_b=*/false, b,
        /*c_mod=*/(short)0, c, /*reuse_a=*/false, /*reuse_b=*/false);
}

// ---------------------------------------------------------------------------
// One 128-row attention problem:
//   S = (Q Kt) [*mask] * scale ; P = softmax_row(S) ; O = P V
// Q/K/V rows r live at   base + r*rowStride   (64 contiguous fp32 each).
// maskRows: nullptr, or mask[b] with row stride 128 (row index = query row).
// ---------------------------------------------------------------------------
__device__ __forceinline__ void attn_core(const float* __restrict__ Qg,
                                          const float* __restrict__ Kg,
                                          const float* __restrict__ Vg,
                                          const float* __restrict__ maskRows,
                                          float* __restrict__ Og,
                                          int base, int rowStride, char* smem) {
    unsigned short* Qs = (unsigned short*)smem;   // [128][64] bf16  16 KB
    unsigned short* Ks = Qs + 128 * 64;           // [128][64] bf16  16 KB
    unsigned short* Vt = Ks + 128 * 64;           // [64][128] bf16 (d-major) 16 KB
    unsigned short* Ps = Vt + 64 * 128;           // [128][128] bf16 probs 32 KB
    float* Srow = (float*)(Ps + 128 * 128);       // [8 waves][16][128] f32 64 KB
    float* invs = Srow + 8 * 16 * 128;            // [128] f32 per-row 1/sum

    const int tid = threadIdx.x;

    // ---- stage Q,K,V: global fp32 -> LDS bf16 (V transposed to d-major) ----
    for (int idx = tid; idx < 128 * 16; idx += 256) {
        int r = idx >> 4;
        int e = (idx & 15) << 2;
        int g = base + r * rowStride + e;
        v4f q = *(const v4f*)(Qg + g);
        v4f k = *(const v4f*)(Kg + g);
        v4f v = *(const v4f*)(Vg + g);
        v4us qb = { f2bf(q[0]), f2bf(q[1]), f2bf(q[2]), f2bf(q[3]) };
        v4us kb = { f2bf(k[0]), f2bf(k[1]), f2bf(k[2]), f2bf(k[3]) };
        *(v4us*)(Qs + r * 64 + e) = qb;
        *(v4us*)(Ks + r * 64 + e) = kb;
        Vt[(e + 0) * 128 + r] = f2bf(v[0]);
        Vt[(e + 1) * 128 + r] = f2bf(v[1]);
        Vt[(e + 2) * 128 + r] = f2bf(v[2]);
        Vt[(e + 3) * 128 + r] = f2bf(v[3]);
    }
    __syncthreads();

    const int lane = tid & 31;
    const int w    = tid >> 5;      // 8 waves, wave w owns rows [16w, 16w+16)
    const int lo   = lane & 15;
    const int hi   = lane >> 4;
    const int rb   = w * 16;
    const float scale = 0.125f;     // 1/sqrt(E=64)

    // ---- S = Q Kt for this wave's 16 rows (8 column tiles, K=64 in 2 steps) ----
    v16bf aq0 = ld_a_frag(Qs, 64, rb, 0,  lo, hi);
    v16bf aq1 = ld_a_frag(Qs, 64, rb, 32, lo, hi);
    float* Sw = Srow + w * (16 * 128);
#pragma unroll
    for (int t = 0; t < 8; ++t) {
        v16bf b0 = ld_b_frag(Ks, 64, t * 16, 0,  lo, hi);
        v16bf b1 = ld_b_frag(Ks, 64, t * 16, 32, lo, hi);
        v8f c = {0.f, 0.f, 0.f, 0.f, 0.f, 0.f, 0.f, 0.f};
        c = wmma_bf16(aq0, b0, c);
        c = wmma_bf16(aq1, b1, c);
        // C layout: VGPR r -> (M = r + 8*hi, N = lo)
#pragma unroll
        for (int r = 0; r < 8; ++r)
            Sw[(r + 8 * hi) * 128 + t * 16 + lo] = c[r];
    }

    // ---- row softmax: 2 lanes per row (lane&15 = row, lane>>4 = col half),
    //      v4f vectorized; max/sum combined across the pair with shfl_xor(16).
    {
        const int row = rb + lo;            // this lane's row
        const int cb  = hi * 64;            // this lane's 64-column half
        const float* s    = Sw + lo * 128 + cb;
        const float* mrow = maskRows ? (maskRows + row * 128 + cb) : nullptr;

        float mx = -3.402823466e38f;
#pragma unroll
        for (int i = 0; i < 16; ++i) {
            v4f v = *(const v4f*)(s + i * 4);
            if (mrow) v *= *(const v4f*)(mrow + i * 4);   // multiplicative mask
            v *= scale;
            mx = fmaxf(mx, fmaxf(fmaxf(v[0], v[1]), fmaxf(v[2], v[3])));
        }
        mx = fmaxf(mx, __shfl_xor(mx, 16, 32));           // combine halves

        float sum = 0.f;
        unsigned short* prow = Ps + row * 128 + cb;
#pragma unroll
        for (int i = 0; i < 16; ++i) {
            v4f v = *(const v4f*)(s + i * 4);
            if (mrow) v *= *(const v4f*)(mrow + i * 4);
            v *= scale;
            float e0 = __expf(v[0] - mx);
            float e1 = __expf(v[1] - mx);
            float e2 = __expf(v[2] - mx);
            float e3 = __expf(v[3] - mx);
            sum += (e0 + e1) + (e2 + e3);
            v4us pb = { f2bf(e0), f2bf(e1), f2bf(e2), f2bf(e3) };
            *(v4us*)(prow + i * 4) = pb;                  // unnormalized bf16 probs
        }
        sum += __shfl_xor(sum, 16, 32);                   // combine halves
        if (hi == 0) invs[row] = 1.0f / sum;
    }
    // Ps/invs producers and consumers are the same wave: DS ops from one wave
    // execute in order, no workgroup barrier needed.

    // ---- O = P V : 4 d-tiles x (K=128 over 4 chunks of 32) ----
    v16bf ap[4];
#pragma unroll
    for (int st = 0; st < 4; ++st)
        ap[st] = ld_a_frag(Ps, 128, rb, st * 32, lo, hi);

#pragma unroll
    for (int dt = 0; dt < 4; ++dt) {
        v8f o = {0.f, 0.f, 0.f, 0.f, 0.f, 0.f, 0.f, 0.f};
#pragma unroll
        for (int st = 0; st < 4; ++st) {
            v16bf bv = ld_b_frag(Vt, 128, dt * 16, st * 32, lo, hi);
            o = wmma_bf16(ap[st], bv, o);
        }
#pragma unroll
        for (int r = 0; r < 8; ++r) {
            int row = rb + r + 8 * hi;
            Og[base + row * rowStride + dt * 16 + lo] = o[r] * invs[row];
        }
    }
}

// ---------------------------------------------------------------------------
// Pass 1: temporal attention per (b,n,h); rows = l, rowStride = H*E = 512.
// ---------------------------------------------------------------------------
__global__ void __launch_bounds__(256)
st_temporal(const float* __restrict__ Q, const float* __restrict__ K,
            const float* __restrict__ V, float* __restrict__ O) {
    extern __shared__ char smem[];
    const int pid = blockIdx.x;          // B*N*H = 2048
    const int h = pid & 7;
    const int n = (pid >> 3) & 127;
    const int b = pid >> 10;
    const int base = (b * 128 + n) * 65536 + h * 64;   // (b,n,l=0,h,e=0)
    attn_core(Q, K, V, nullptr, O, base, 512, smem);
}

// ---------------------------------------------------------------------------
// Pass 2: spatial attention per (b,l,h); rows = n, rowStride = L*H*E = 65536.
// V source is the temporal output V_t (fp32, same layout). Mask applied.
// ---------------------------------------------------------------------------
__global__ void __launch_bounds__(256)
st_spatial(const float* __restrict__ Q, const float* __restrict__ K,
           const float* __restrict__ Vt_in, const float* __restrict__ mask,
           float* __restrict__ O) {
    extern __shared__ char smem[];
    const int pid = blockIdx.x;          // B*L*H = 2048
    const int h = pid & 7;
    const int l = (pid >> 3) & 127;
    const int b = pid >> 10;
    const int base = b * (128 * 128 * 8 * 64) + (l * 8 + h) * 64;  // (b,n=0,l,h,0)
    attn_core(Q, K, Vt_in, mask + b * (128 * 128), O, base, 65536, smem);
}

extern "C" void kernel_launch(void* const* d_in, const int* in_sizes, int n_in,
                              void* d_out, int out_size, void* d_ws, size_t ws_size,
                              hipStream_t stream) {
    (void)in_sizes; (void)n_in; (void)out_size; (void)ws_size;
    const float* Q    = (const float*)d_in[0];
    const float* K    = (const float*)d_in[1];
    const float* V    = (const float*)d_in[2];
    const float* mask = (const float*)d_in[3];
    float* out  = (float*)d_out;
    float* v_t  = (float*)d_ws;   // [B,N,L,H,E] fp32 intermediate = 64 MB

    // dynamic LDS: Qs+Ks+Vt (3*16KB bf16) + Ps (32KB bf16) + Srow (64KB f32) + invs
    const size_t smem = (size_t)(3 * 128 * 64 * 2 + 128 * 128 * 2 +
                                 8 * 16 * 128 * 4 + 128 * 4);

    hipLaunchKernelGGL(st_temporal, dim3(2048), dim3(256), smem, stream,
                       Q, K, V, v_t);
    hipLaunchKernelGGL(st_spatial, dim3(2048), dim3(256), smem, stream,
                       Q, K, v_t, mask, out);
}
